// LocalAttention_66202625901147
// MI455X (gfx1250) — matrix operand order repaired and checked by reading, not compiled, not run on hardware
//
#include <hip/hip_runtime.h>
#include <hip/hip_bf16.h>
#include <cstdint>
#include <cstddef>

// ---------------------------------------------------------------------------
// LocalAttention for MI455X (gfx1250).
// Compute budget: 4 GEMMs of [32768x1024]@[1024x1024] = 275 GFLOP (dominant),
// attention itself ~1 GFLOP, HBM floor ~420MB (~18us @ 23.3 TB/s).
// => bf16 WMMA (v_wmma_f32_16x16x32_bf16, fp32 accum) for the GEMMs,
//    async global->LDS (ASYNCcnt, GVS saddr form) double-buffered pipeline
//    with ONE barrier per K-step, VALU for the tiny per-position softmax.
// ---------------------------------------------------------------------------

typedef __attribute__((ext_vector_type(16))) __bf16 v16bf;
typedef __attribute__((ext_vector_type(8)))  float  v8f;

#define L_DIM   8
#define S_DIM   4096
#define DM      1024
#define H_HEADS 16
#define HD      64
#define WND     128
#define MROWS   (L_DIM * S_DIM)          // 32768 GEMM rows

// ---- CDNA5 async global->LDS helpers (inline asm; tracked by ASYNCcnt) ----
// GVS addressing: mem = SGPR64 base + VGPR32 byte offset. The base is uniform
// per K-step (scalar add), the per-lane offset is loop-invariant.
__device__ __forceinline__ void async_b128_s(uint32_t lds_off, uint32_t voff,
                                             const void* sbase) {
  asm volatile("global_load_async_to_lds_b128 %0, %1, %2"
               :: "v"(lds_off), "v"(voff),
                  "s"((unsigned long long)(uintptr_t)sbase)
               : "memory");
}
template <int N>
__device__ __forceinline__ void wait_async() {
  asm volatile("s_wait_asynccnt %0" :: "n"(N) : "memory");
}

union BF16x16 { v16bf v; uint4 q[2]; };

// A fragment: 16x32 (MxK) bf16, row-major LDS tile with row stride 32 elems.
// Lane 0-15 -> M=lane, K halves {0..7,16..23}; lane 16-31 -> M=lane-16,
// K halves {8..15,24..31}. Each half is 8 contiguous bf16 = one b128 read.
__device__ __forceinline__ v16bf load_frag_a(const __bf16* sA, int tileRow, int lane) {
  const int row  = tileRow + (lane & 15);
  const int half = (lane >> 4);               // 0 or 1
  const uint4* p = (const uint4*)(sA + row * 32);
  BF16x16 f;
  f.q[0] = p[half];                           // K = half*8 .. +7
  f.q[1] = p[2 + half];                       // K = 16+half*8 .. +7
  return f.v;
}

// B fragment: 32x16 (KxN) bf16. LDS tile holds sB[n][k] (row stride 32).
// Lane: N = lane&15, K halves {0..7,16..23} (lane<16) / {8..15,24..31} (lane>=16).
__device__ __forceinline__ v16bf load_frag_b(const __bf16* sB, int tileCol, int lane) {
  const int n     = tileCol + (lane & 15);
  const int khalf = (lane >> 4);              // 0 or 1
  const uint4* p  = (const uint4*)(sB + n * 32);
  BF16x16 f;
  f.q[0] = p[khalf];
  f.q[1] = p[2 + khalf];
  return f.v;
}

// ---------------------------------------------------------------------------
// GEMM: Y[M=32768, N=1024] = A[M,1024] @ W[1024,1024]^T + bias*bias_scale
// Macro tile 128x128, 8 waves (4x2) each 32x64 (2x4 WMMA tiles), K step 32,
// async double-buffer, single barrier per K-step, rolled loop (stable accs).
// ---------------------------------------------------------------------------
__global__ __launch_bounds__(256)
void gemm_bf16_wmma(const __bf16* __restrict__ A, const __bf16* __restrict__ W,
                    const float* __restrict__ bias, float bias_scale,
                    __bf16* __restrict__ Ybf, float* __restrict__ Yf) {
  __shared__ __align__(16) __bf16 sA[2][128 * 32];   // 2 x 8KB
  __shared__ __align__(16) __bf16 sB[2][128 * 32];   // 2 x 8KB

  const int tid  = threadIdx.x;
  const int lane = tid & 31;
  const int wave = tid >> 5;
  const int wm   = wave >> 1;                 // 0..3  (rows of 32)
  const int wn   = wave & 1;                  // 0..1  (cols of 64)
  const int rowBase = blockIdx.y * 128;
  const int colBase = blockIdx.x * 128;

  // Loop-invariant per-lane pieces. 128 rows x 4 b128-chunks per tile =
  // 512 chunks; this thread owns chunks tid and tid+256 in each tile.
  // A and B tiles share the same chunk->offset mapping.
  const int r0 = tid >> 2,         c0 = tid & 3;
  const int r1 = (tid + 256) >> 2, c1 = (tid + 256) & 3;
  const uint32_t gv0 = (uint32_t)((r0 * DM + c0 * 8) * 2);   // global byte offs
  const uint32_t gv1 = (uint32_t)((r1 * DM + c1 * 8) * 2);
  const uint32_t ldsA0 = (uint32_t)(uintptr_t)(&sA[0][r0 * 32 + c0 * 8]);
  const uint32_t ldsA1 = (uint32_t)(uintptr_t)(&sA[0][r1 * 32 + c1 * 8]);
  const uint32_t ldsB0 = (uint32_t)(uintptr_t)(&sB[0][r0 * 32 + c0 * 8]);
  const uint32_t ldsB1 = (uint32_t)(uintptr_t)(&sB[0][r1 * 32 + c1 * 8]);

  v8f acc[2][4] = {};

  {                                           // prefetch K-step 0 into buf 0
    const __bf16* Ag = A + (size_t)rowBase * DM;
    const __bf16* Wg = W + (size_t)colBase * DM;
    async_b128_s(ldsA0, gv0, Ag);
    async_b128_s(ldsA1, gv1, Ag);
    async_b128_s(ldsB0, gv0, Wg);
    async_b128_s(ldsB1, gv1, Wg);
  }

  const int KT = DM / 32;                     // 32 K-steps
#pragma unroll 1
  for (int kt = 0; kt < KT; ++kt) {
    const int buf = kt & 1;
    wait_async<0>();                          // this wave's current tiles done
    __syncthreads();                          // all waves' tiles done + prev
                                              // buffer fully consumed
    if (kt + 1 < KT) {                        // prefetch next (overlaps MMAs)
      const __bf16* Ag = A + (size_t)rowBase * DM + (kt + 1) * 32;
      const __bf16* Wg = W + (size_t)colBase * DM + (kt + 1) * 32;
      const uint32_t bo = (uint32_t)(buf ^ 1) * (128 * 32 * 2);
      async_b128_s(ldsA0 + bo, gv0, Ag);
      async_b128_s(ldsA1 + bo, gv1, Ag);
      async_b128_s(ldsB0 + bo, gv0, Wg);
      async_b128_s(ldsB1 + bo, gv1, Wg);
    }
#pragma unroll
    for (int mi = 0; mi < 2; ++mi) {
      const v16bf a = load_frag_a(sA[buf], wm * 32 + mi * 16, lane);
#pragma unroll
      for (int ni = 0; ni < 4; ++ni) {
        const v16bf b = load_frag_b(sB[buf], wn * 64 + ni * 16, lane);
        acc[mi][ni] = __builtin_amdgcn_wmma_f32_16x16x32_bf16(
            false, a, false, b, (short)0, acc[mi][ni], false, false);
      }
    }
  }

  // Epilogue. C layout: VGPR r -> M = r + (lane>=16 ? 8:0); N = lane&15.
#pragma unroll
  for (int mi = 0; mi < 2; ++mi) {
#pragma unroll
    for (int ni = 0; ni < 4; ++ni) {
      const int col = colBase + wn * 64 + ni * 16 + (lane & 15);
      const float badd = bias[col] * bias_scale;
      const int rbase = rowBase + wm * 32 + mi * 16 + ((lane >> 4) << 3);
#pragma unroll
      for (int r = 0; r < 8; ++r) {
        const float v = acc[mi][ni][r] + badd;
        const size_t off = (size_t)(rbase + r) * DM + col;
        if (Yf) Yf[off] = v;
        else    Ybf[off] = (__bf16)v;
      }
    }
  }
}

// ---------------------------------------------------------------------------
// Sliding-window mean over S (value -> bf16 kv). Thread per (b, d, s-chunk):
// 16 chunks of 256 steps -> 131072 threads.
// ---------------------------------------------------------------------------
__global__ void pool_kernel(const float* __restrict__ value, __bf16* __restrict__ kvb) {
  const int idx = blockIdx.x * blockDim.x + threadIdx.x;   // 0..131071
  const int d = idx & 1023;
  const int t = idx >> 10;                    // 0..127
  const int b = t >> 4;                       // 0..7
  const int s0 = (t & 15) * 256;              // chunk start
  const float* base = value + ((size_t)b * S_DIM) * DM + d;

  int l = s0 - WND; if (l < 0) l = 0;
  int r = s0 + WND + 1; if (r > S_DIM) r = S_DIM;
  float run = 0.f;
  for (int s = l; s < r; ++s) run += base[(size_t)s * DM];
  int cnt = r - l;

  for (int s = s0; s < s0 + 256; ++s) {
    kvb[((size_t)b * S_DIM + s) * DM + d] = (__bf16)(run / (float)cnt);
    const int add = s + WND + 1;
    if (add < S_DIM) { run += base[(size_t)add * DM]; ++cnt; }
    const int rem = s - WND;
    if (rem >= 0)    { run -= base[(size_t)rem * DM]; --cnt; }
  }
}

// fp32 -> bf16 with scale, 4 elements per thread (float4 in, 4x bf16 out).
__global__ void cvt_bf16_kernel(const float* __restrict__ x, __bf16* __restrict__ y,
                                float scale, int n4) {
  const int i = blockIdx.x * blockDim.x + threadIdx.x;
  if (i < n4) {
    const float4 v = ((const float4*)x)[i];
    union { __bf16 h[4]; uint2 u; } o;
    o.h[0] = (__bf16)(v.x * scale);
    o.h[1] = (__bf16)(v.y * scale);
    o.h[2] = (__bf16)(v.z * scale);
    o.h[3] = (__bf16)(v.w * scale);
    ((uint2*)y)[i] = o.u;
  }
}

// ---------------------------------------------------------------------------
// Attention: per n (block), thread = (head h, row l). 8x8 scores over hd=64,
// fp32 softmax, bf16 context, head-averaged weights [n][l][m].
// ---------------------------------------------------------------------------
__global__ __launch_bounds__(128)
void attn_kernel(const __bf16* __restrict__ qp, const __bf16* __restrict__ kp,
                 const __bf16* __restrict__ vp, __bf16* __restrict__ ctx,
                 float* __restrict__ wout) {
  __shared__ __align__(16) __bf16 sq[8][DM];
  __shared__ __align__(16) __bf16 sk[8][DM];
  __shared__ __align__(16) __bf16 sv[8][DM];
  __shared__ float sattn[H_HEADS][8][8];

  const int n = blockIdx.x;
  const int tid = threadIdx.x;

  // Stage 8 rows x 1024 bf16 per tensor (each row 128 b128 chunks).
#pragma unroll
  for (int i = 0; i < 8; ++i) {
    const int t = tid + i * 128;              // 0..1023
    const int l = t >> 7, c = t & 127;
    const size_t rowOff = ((size_t)l * S_DIM + n) * DM;
    ((uint4*)sq[l])[c] = ((const uint4*)(qp + rowOff))[c];
    ((uint4*)sk[l])[c] = ((const uint4*)(kp + rowOff))[c];
    ((uint4*)sv[l])[c] = ((const uint4*)(vp + rowOff))[c];
  }
  __syncthreads();

  const int h = tid >> 3, l = tid & 7;
  const __bf16* qrow = &sq[l][h * HD];

  float sc[8];
#pragma unroll
  for (int m = 0; m < 8; ++m) {
    const __bf16* krow = &sk[m][h * HD];
    float s = 0.f;
    for (int d = 0; d < HD; ++d) s += (float)qrow[d] * (float)krow[d];
    sc[m] = s;                                // q already carries 1/sqrt(hd)
  }
  float mx = sc[0];
#pragma unroll
  for (int m = 1; m < 8; ++m) mx = fmaxf(mx, sc[m]);
  float sum = 0.f;
#pragma unroll
  for (int m = 0; m < 8; ++m) { sc[m] = __expf(sc[m] - mx); sum += sc[m]; }
  const float inv = 1.f / sum;
#pragma unroll
  for (int m = 0; m < 8; ++m) { sc[m] *= inv; sattn[h][l][m] = sc[m]; }

  // context: out[l, n, h, :] = sum_m p[m] * v[m, n, h, :]
  for (int d = 0; d < HD; ++d) {
    float a = 0.f;
#pragma unroll
    for (int m = 0; m < 8; ++m) a += sc[m] * (float)sv[m][h * HD + d];
    ctx[((size_t)l * S_DIM + n) * DM + h * HD + d] = (__bf16)a;
  }
  __syncthreads();

  if (tid < 64) {
    const int ll = tid >> 3, mm = tid & 7;
    float a = 0.f;
#pragma unroll
    for (int hh = 0; hh < H_HEADS; ++hh) a += sattn[hh][ll][mm];
    wout[(size_t)n * 64 + ll * 8 + mm] = a * (1.f / 16.f);
  }
}

// ---------------------------------------------------------------------------
extern "C" void kernel_launch(void* const* d_in, const int* in_sizes, int n_in,
                              void* d_out, int out_size, void* d_ws, size_t ws_size,
                              hipStream_t stream) {
  const float* query = (const float*)d_in[0];
  // d_in[1] == key: dead due to the reference's aliasing bug.
  const float* value = (const float*)d_in[2];
  const float* in_w  = (const float*)d_in[3];
  const float* in_b  = (const float*)d_in[4];
  const float* out_w = (const float*)d_in[5];
  const float* out_b = (const float*)d_in[6];

  float* out     = (float*)d_out;                       // [8,4096,1024]
  float* weights = out + (size_t)L_DIM * S_DIM * DM;    // [4096,8,8]

  const size_t NE = (size_t)MROWS * DM;                 // 33,554,432 elems
  char* ws = (char*)d_ws;
  __bf16* xq  = (__bf16*)ws; ws += NE * 2;              // scaled query
  __bf16* kvb = (__bf16*)ws; ws += NE * 2;              // pooled value
  __bf16* wb  = (__bf16*)ws; ws += (size_t)4 * 1024 * 1024 * 2;  // Wq|Wk|Wv|Wo
  __bf16* qp  = (__bf16*)ws; ws += NE * 2;
  __bf16* kp  = (__bf16*)ws; ws += NE * 2;
  __bf16* vp  = (__bf16*)ws; ws += NE * 2;
  __bf16* cx  = (__bf16*)ws; ws += NE * 2;

  const float qscale = 0.125f;                          // 1/sqrt(64)

  // Converts + pooling
  cvt_bf16_kernel<<<(int)(NE / 4 / 256), 256, 0, stream>>>(query, xq, qscale, (int)(NE / 4));
  cvt_bf16_kernel<<<(3 * 1024 * 1024 / 4) / 256, 256, 0, stream>>>(in_w, wb, 1.f, 3 * 1024 * 1024 / 4);
  cvt_bf16_kernel<<<(1024 * 1024 / 4) / 256, 256, 0, stream>>>(out_w, wb + (size_t)3 * 1024 * 1024, 1.f, 1024 * 1024 / 4);
  pool_kernel<<<512, 256, 0, stream>>>(value, kvb);

  // Projections (bf16 WMMA, fp32 accumulate)
  dim3 ggrid(DM / 128, MROWS / 128);                    // 8 x 256 = 2048 blocks
  gemm_bf16_wmma<<<ggrid, 256, 0, stream>>>(xq,  wb,                           in_b,        qscale, qp, nullptr);
  gemm_bf16_wmma<<<ggrid, 256, 0, stream>>>(kvb, wb + (size_t)1024 * 1024,     in_b + 1024, 1.f,    kp, nullptr);
  gemm_bf16_wmma<<<ggrid, 256, 0, stream>>>(kvb, wb + (size_t)2 * 1024 * 1024, in_b + 2048, 1.f,    vp, nullptr);

  // Tiny per-position attention + averaged weights
  attn_kernel<<<S_DIM, 128, 0, stream>>>(qp, kp, vp, cx, weights);

  // Output projection straight to fp32 d_out
  gemm_bf16_wmma<<<ggrid, 256, 0, stream>>>(cx, wb + (size_t)3 * 1024 * 1024, out_b, 1.f, nullptr, out);
}
